// GNNImitator_19378892439725
// MI455X (gfx1250) — matrix operand order repaired and checked by reading, not compiled, not run on hardware
//
#include <hip/hip_runtime.h>
#include <hip/hip_bf16.h>

#define NN      50000
#define EE      800000
#define TOTE    (EE + NN)
#define INC     128
#define HIDC    128
#define HEADS   4
#define DHD     32
#define OUTC    32
#define BATCH   1024
#define NEGSLOPE 0.2f

typedef __bf16 v16bf __attribute__((ext_vector_type(16)));
typedef float  v8f   __attribute__((ext_vector_type(8)));

// ---------------- conversion / transpose ----------------
__global__ void k_f32_to_bf16(const float* __restrict__ in, __bf16* __restrict__ out, long n) {
  long i = (long)blockIdx.x * blockDim.x + threadIdx.x;
  if (i < n) out[i] = (__bf16)in[i];
}

// Wt[n*K + k] = (bf16) W[k*N + n]   (W is [K,N] row-major -> Wt is [N,K] row-major)
__global__ void k_transpose_bf16(const float* __restrict__ W, __bf16* __restrict__ Wt, int K, int N) {
  int i = blockIdx.x * blockDim.x + threadIdx.x;
  if (i >= K * N) return;
  int n = i / K;
  int k = i - n * K;
  Wt[i] = (__bf16)W[(long)k * N + n];
}

// ---------------- WMMA GEMM ----------------
// C[M,N] = act(A[M,K] @ Bt[N,K]^T + bias).
// One wave32 owns a 16 x (16*NTILES) output strip: A fragment loaded once per K-step,
// reused for NTILES wmmas. Bt (the whole weight matrix) is staged in LDS by the block,
// padded by 8 bf16 per row (16B) to skew LDS banks across the 16 column-lanes.
//
// A fragment (16x32 bf16): lane l -> row mt*16+(l&15), elems 0..7 = K (l>>4)*16 + 0..7,
//                          elems 8..15 = K (l>>4)*16 + 8..15  (two 16B global loads)
// B fragment (32x16 bf16): lane l -> col nt*16+(l&15), elems 0..15 = K (l>>4)*16 + 0..15
//                          (two 16B LDS loads)
// C fragment (16x16 f32):  lane l -> col (l&15), rows (l>>4)*8 + r
template<int NTILES>
__global__ __launch_bounds__(256)
void k_wmma_gemm(const __bf16* __restrict__ A, const __bf16* __restrict__ Bt,
                 const float* __restrict__ bias,
                 float* __restrict__ Cf, __bf16* __restrict__ Cb,
                 int M, int K, int relu)
{
  extern __shared__ char smem_raw[];
  __bf16* sB = (__bf16*)smem_raw;
  const int N  = NTILES * 16;
  const int SK = K + 8;                       // padded LDS row stride (bank skew)

  // cooperative stage of Bt[N,K] -> LDS (K is a multiple of 8, chunks stay in-row)
  for (int e = threadIdx.x * 8; e < N * K; e += blockDim.x * 8) {
    int n = e / K;
    int k = e - n * K;
    *(uint4*)(sB + n * SK + k) = *(const uint4*)(Bt + e);
  }
  __syncthreads();

  int lane = threadIdx.x & 31;
  int mt   = blockIdx.x * (blockDim.x >> 5) + (threadIdx.x >> 5);
  if (mt * 16 >= M) return;

  int half = lane >> 4;     // 0 or 1
  int l15  = lane & 15;
  const __bf16* ap = A + (long)(mt * 16 + l15) * K + half * 16;
  const __bf16* bbase = sB + (long)l15 * SK + half * 16;

  v8f acc[NTILES];
#pragma unroll
  for (int t = 0; t < NTILES; ++t) acc[t] = {};

  for (int k0 = 0; k0 < K; k0 += 32) {
    v16bf a;
    ((uint4*)&a)[0] = *(const uint4*)(ap + k0);        // K = base .. base+7
    ((uint4*)&a)[1] = *(const uint4*)(ap + k0 + 8);    // K = base+8 .. base+15
#pragma unroll
    for (int nt = 0; nt < NTILES; ++nt) {
      v16bf b;
      const __bf16* bp = bbase + nt * 16 * SK + k0;
      ((uint4*)&b)[0] = *(const uint4*)(bp);           // K = base .. base+15
      ((uint4*)&b)[1] = *(const uint4*)(bp + 8);
      acc[nt] = __builtin_amdgcn_wmma_f32_16x16x32_bf16(
          /*neg_a=*/false, a, /*neg_b=*/false, b,
          /*c_mod=*/(short)0, acc[nt], /*reuse_a=*/false, /*reuse_b=*/false);
    }
  }

  int mrow = mt * 16 + half * 8;
#pragma unroll
  for (int nt = 0; nt < NTILES; ++nt) {
    int ccol = nt * 16 + l15;
    float bv = bias ? bias[ccol] : 0.0f;
#pragma unroll
    for (int r = 0; r < 8; ++r) {
      float v = acc[nt][r] + bv;
      if (relu) v = v > 0.0f ? v : 0.0f;
      long off = (long)(mrow + r) * N + ccol;
      if (Cf) Cf[off] = v;
      if (Cb) Cb[off] = (__bf16)v;
    }
  }
}

// ---------------- per-node attention logits ----------------
__global__ void k_node_e(const float* __restrict__ xp,
                         const float* __restrict__ a_src, const float* __restrict__ a_dst,
                         float* __restrict__ es, float* __restrict__ ed)
{
  int i = blockIdx.x * blockDim.x + threadIdx.x;   // node*HEADS + h
  if (i >= NN * HEADS) return;
  int n = i >> 2, h = i & 3;
  const float* v = xp + (long)n * HIDC + h * DHD;
  float ss = 0.0f, sd = 0.0f;
#pragma unroll
  for (int k = 0; k < DHD; ++k) {
    float t = v[k];
    ss += t * a_src[h * DHD + k];
    sd += t * a_dst[h * DHD + k];
  }
  es[i] = ss;
  ed[i] = sd;
}

// ---------------- edge softmax passes ----------------
__global__ void k_edge_pass1(const int* __restrict__ ei,
                             const float* __restrict__ es, const float* __restrict__ ed,
                             float* __restrict__ ew, unsigned* __restrict__ mkey)
{
  int i = blockIdx.x * blockDim.x + threadIdx.x;
  if (i >= TOTE) return;
  int s, d;
  if (i < EE) { s = ei[i]; d = ei[EE + i]; } else { s = i - EE; d = s; }
#pragma unroll
  for (int h = 0; h < HEADS; ++h) {
    float v = es[s * HEADS + h] + ed[d * HEADS + h];
    v = (v > 0.0f) ? v : NEGSLOPE * v;               // leaky relu
    ew[(long)i * HEADS + h] = v;
    unsigned u   = __float_as_uint(v);
    unsigned key = (u & 0x80000000u) ? ~u : (u | 0x80000000u);  // monotonic order key
    atomicMax(mkey + d * HEADS + h, key);
  }
}

__global__ void k_edge_pass2(const int* __restrict__ ei,
                             const unsigned* __restrict__ mkey,
                             float* __restrict__ ew, float* __restrict__ zsum)
{
  int i = blockIdx.x * blockDim.x + threadIdx.x;
  if (i >= TOTE) return;
  int d = (i < EE) ? ei[EE + i] : (i - EE);
#pragma unroll
  for (int h = 0; h < HEADS; ++h) {
    unsigned key = mkey[d * HEADS + h];
    unsigned u   = (key & 0x80000000u) ? (key & 0x7FFFFFFFu) : ~key;
    float m  = __uint_as_float(u);
    float wv = __expf(ew[(long)i * HEADS + h] - m);
    ew[(long)i * HEADS + h] = wv;
    atomicAdd(zsum + d * HEADS + h, wv);
  }
}

// one wave32 per edge; lane handles 4 contiguous feature dims (same head)
__global__ void k_edge_pass3(const int* __restrict__ ei,
                             const float* __restrict__ ew, const float* __restrict__ zsum,
                             const float* __restrict__ xp, float* __restrict__ agg)
{
  long t = (long)blockIdx.x * blockDim.x + threadIdx.x;
  int wid  = (int)(t >> 5);
  int lane = (int)(t & 31);
  if (wid >= TOTE) return;
  int s, d;
  if (wid < EE) { s = ei[wid]; d = ei[EE + wid]; } else { s = wid - EE; d = s; }
  int k = lane * 4;
  int h = k >> 5;
  float alpha = ew[(long)wid * HEADS + h] / zsum[d * HEADS + h];
  const float4 xv = *(const float4*)(xp + (long)s * HIDC + k);
  float* ap = agg + (long)d * HIDC + k;
  atomicAdd(ap + 0, alpha * xv.x);
  atomicAdd(ap + 1, alpha * xv.y);
  atomicAdd(ap + 2, alpha * xv.z);
  atomicAdd(ap + 3, alpha * xv.w);
}

// ---------------- epilogues ----------------
__global__ void k_bias_relu_bf16(const float* __restrict__ agg, const float* __restrict__ bias,
                                 __bf16* __restrict__ hb)
{
  long i = (long)blockIdx.x * blockDim.x + threadIdx.x;
  if (i >= (long)NN * HIDC) return;
  float v = agg[i] + bias[i & (HIDC - 1)];
  v = v > 0.0f ? v : 0.0f;
  hb[i] = (__bf16)v;
}

__global__ void k_final_combine(const float* __restrict__ agg, const float* __restrict__ bias,
                                const float* __restrict__ resid, float* __restrict__ hf)
{
  long i = (long)blockIdx.x * blockDim.x + threadIdx.x;
  if (i >= (long)NN * HIDC) return;
  float v = agg[i] + bias[i & (HIDC - 1)];
  v = v > 0.0f ? v : 0.0f;
  hf[i] = v + resid[i];
}

__global__ void k_gather_bf16(const float* __restrict__ hf, const int* __restrict__ idx,
                              __bf16* __restrict__ emb)
{
  int i = blockIdx.x * blockDim.x + threadIdx.x;
  if (i >= BATCH * HIDC) return;
  int b = i >> 7, k = i & 127;
  emb[i] = (__bf16)hf[(long)idx[b] * HIDC + k];
}

// ---------------- host ----------------
extern "C" void kernel_launch(void* const* d_in, const int* in_sizes, int n_in,
                              void* d_out, int out_size, void* d_ws, size_t ws_size,
                              hipStream_t stream) {
  (void)in_sizes; (void)n_in; (void)out_size; (void)ws_size;

  const float* x     = (const float*)d_in[0];
  const int*   cni   = (const int*)d_in[1];
  const int*   ei    = (const int*)d_in[2];
  const float* W_in  = (const float*)d_in[3];
  const float* b_in  = (const float*)d_in[4];
  const float* W_res = (const float*)d_in[5];
  const float* b_res = (const float*)d_in[6];
  const float* W_g1  = (const float*)d_in[7];
  const float* a_s1  = (const float*)d_in[8];
  const float* a_d1  = (const float*)d_in[9];
  const float* b_g1  = (const float*)d_in[10];
  const float* W_g2  = (const float*)d_in[11];
  const float* a_s2  = (const float*)d_in[12];
  const float* a_d2  = (const float*)d_in[13];
  const float* b_g2  = (const float*)d_in[14];
  const float* W_o1  = (const float*)d_in[15];
  const float* b_o1  = (const float*)d_in[16];
  const float* W_o2  = (const float*)d_in[17];
  const float* b_o2  = (const float*)d_in[18];
  float* out = (float*)d_out;

  char* wsp = (char*)d_ws;
  auto alloc = [&](size_t bytes) -> char* {
    char* p = wsp;
    wsp += (bytes + 255) & ~(size_t)255;
    return p;
  };
  __bf16*   xb    = (__bf16*)  alloc((size_t)NN * INC * 2);      // x in bf16
  __bf16*   hb    = (__bf16*)  alloc((size_t)NN * HIDC * 2);     // current activation (bf16)
  float*    resid = (float*)   alloc((size_t)NN * HIDC * 4);
  float*    xp    = (float*)   alloc((size_t)NN * HIDC * 4);     // per-layer transformed feats (later hfinal)
  float*    agg   = (float*)   alloc((size_t)NN * HIDC * 4);     // message accumulator
  float*    ew    = (float*)   alloc((size_t)TOTE * HEADS * 4);  // per-edge logits/weights
  float*    es    = (float*)   alloc((size_t)NN * HEADS * 4);
  float*    ed    = (float*)   alloc((size_t)NN * HEADS * 4);
  unsigned* mkey  = (unsigned*)alloc((size_t)NN * HEADS * 4);
  float*    zsum  = (float*)   alloc((size_t)NN * HEADS * 4);
  __bf16*   WtIn  = (__bf16*)  alloc((size_t)HIDC * INC * 2);
  __bf16*   WtRes = (__bf16*)  alloc((size_t)HIDC * HIDC * 2);
  __bf16*   WtG1  = (__bf16*)  alloc((size_t)HIDC * HIDC * 2);
  __bf16*   WtG2  = (__bf16*)  alloc((size_t)HIDC * HIDC * 2);
  __bf16*   WtO1  = (__bf16*)  alloc((size_t)64 * HIDC * 2);
  __bf16*   WtO2  = (__bf16*)  alloc((size_t)OUTC * 64 * 2);
  __bf16*   embb  = (__bf16*)  alloc((size_t)BATCH * HIDC * 2);
  __bf16*   m1b   = (__bf16*)  alloc((size_t)BATCH * 64 * 2);

  auto gemm = [&](const __bf16* A, const __bf16* Bt, const float* bias,
                  float* Cf, __bf16* Cb, int M, int N, int K, int relu) {
    int mtiles = M / 16;
    int blocks = (mtiles + 7) / 8;               // 8 waves (256 threads) per block, 1 M-tile/wave
    size_t shmem = (size_t)N * (K + 8) * 2;      // padded Bt staging
    switch (N >> 4) {
      case 8: k_wmma_gemm<8><<<blocks, 256, shmem, stream>>>(A, Bt, bias, Cf, Cb, M, K, relu); break;
      case 4: k_wmma_gemm<4><<<blocks, 256, shmem, stream>>>(A, Bt, bias, Cf, Cb, M, K, relu); break;
      default: k_wmma_gemm<2><<<blocks, 256, shmem, stream>>>(A, Bt, bias, Cf, Cb, M, K, relu); break;
    }
  };

  // --- precision conversion + weight transposition ---
  {
    long n = (long)NN * INC;
    k_f32_to_bf16<<<(int)((n + 255) / 256), 256, 0, stream>>>(x, xb, n);
  }
  k_transpose_bf16<<<(INC  * HIDC + 255) / 256, 256, 0, stream>>>(W_in,  WtIn,  INC,  HIDC);
  k_transpose_bf16<<<(HIDC * HIDC + 255) / 256, 256, 0, stream>>>(W_res, WtRes, HIDC, HIDC);
  k_transpose_bf16<<<(HIDC * HIDC + 255) / 256, 256, 0, stream>>>(W_g1,  WtG1,  HIDC, HIDC);
  k_transpose_bf16<<<(HIDC * HIDC + 255) / 256, 256, 0, stream>>>(W_g2,  WtG2,  HIDC, HIDC);
  k_transpose_bf16<<<(HIDC * 64   + 255) / 256, 256, 0, stream>>>(W_o1,  WtO1,  HIDC, 64);
  k_transpose_bf16<<<(64   * OUTC + 255) / 256, 256, 0, stream>>>(W_o2,  WtO2,  64,   OUTC);

  // --- h = relu(x @ W_in + b_in) ; residual = h @ W_res + b_res ---
  gemm(xb, WtIn,  b_in,  nullptr, hb, NN, HIDC, INC, 1);
  gemm(hb, WtRes, b_res, resid, nullptr, NN, HIDC, HIDC, 0);

  const int eb  = (TOTE + 255) / 256;
  const int e3b = (int)(((long)TOTE * 32 + 255) / 256);
  const int nb  = (NN * HEADS + 255) / 256;
  const long nh = (long)NN * HIDC;
  const int nhb = (int)((nh + 255) / 256);

  // --- GAT layer 1 ---
  gemm(hb, WtG1, nullptr, xp, nullptr, NN, HIDC, HIDC, 0);
  k_node_e<<<nb, 256, 0, stream>>>(xp, a_s1, a_d1, es, ed);
  hipMemsetAsync(mkey, 0, (size_t)NN * HEADS * 4, stream);
  hipMemsetAsync(zsum, 0, (size_t)NN * HEADS * 4, stream);
  hipMemsetAsync(agg,  0, (size_t)NN * HIDC * 4, stream);
  k_edge_pass1<<<eb, 256, 0, stream>>>(ei, es, ed, ew, mkey);
  k_edge_pass2<<<eb, 256, 0, stream>>>(ei, mkey, ew, zsum);
  k_edge_pass3<<<e3b, 256, 0, stream>>>(ei, ew, zsum, xp, agg);
  k_bias_relu_bf16<<<nhb, 256, 0, stream>>>(agg, b_g1, hb);

  // --- GAT layer 2 ---
  gemm(hb, WtG2, nullptr, xp, nullptr, NN, HIDC, HIDC, 0);
  k_node_e<<<nb, 256, 0, stream>>>(xp, a_s2, a_d2, es, ed);
  hipMemsetAsync(mkey, 0, (size_t)NN * HEADS * 4, stream);
  hipMemsetAsync(zsum, 0, (size_t)NN * HEADS * 4, stream);
  hipMemsetAsync(agg,  0, (size_t)NN * HIDC * 4, stream);
  k_edge_pass1<<<eb, 256, 0, stream>>>(ei, es, ed, ew, mkey);
  k_edge_pass2<<<eb, 256, 0, stream>>>(ei, mkey, ew, zsum);
  k_edge_pass3<<<e3b, 256, 0, stream>>>(ei, ew, zsum, xp, agg);
  k_final_combine<<<nhb, 256, 0, stream>>>(agg, b_g2, resid, xp);   // xp now holds final node features

  // --- output MLP on gathered nodes ---
  k_gather_bf16<<<(BATCH * HIDC + 255) / 256, 256, 0, stream>>>(xp, cni, embb);
  gemm(embb, WtO1, b_o1, nullptr, m1b, BATCH, 64, HIDC, 1);
  gemm(m1b,  WtO2, b_o2, out, nullptr, BATCH, OUTC, 64, 0);
}
